// GRUEncoder_472446402808
// MI455X (gfx1250) — matrix operand order repaired and checked by reading, not compiled, not run on hardware
//
#include <hip/hip_runtime.h>
#include <hip/hip_bf16.h>
#include <math.h>
#include <stdint.h>

// Problem constants (fixed by the reference): V=32000, H=1024, B=64, S=512
#define H_DIM 1024
#define B_DIM 64
#define S_DIM 512
#define K_DIM 1024          // GEMM K (= H)
#define N_DIM 3072          // GEMM N (= 3H)

// persistent recurrence kernel geometry
#define NWG_STEP 16         // workgroups; each owns H/16 = 64 h-columns
#define COLS_WG  64

typedef __attribute__((ext_vector_type(16))) __bf16 v16bf;
typedef __attribute__((ext_vector_type(8)))  __bf16 v8bf;
typedef __attribute__((ext_vector_type(4)))  __bf16 v4bf;
typedef __attribute__((ext_vector_type(8)))  float  v8f;
typedef __attribute__((ext_vector_type(4)))  float  v4f;

static __device__ __forceinline__ v16bf cat8(v8bf lo, v8bf hi) {
  return __builtin_shufflevector(lo, hi, 0,1,2,3,4,5,6,7,8,9,10,11,12,13,14,15);
}

// s_wait_asynccnt: builtin when available, inline asm otherwise
#if __has_builtin(__builtin_amdgcn_s_wait_asynccnt)
#define ASYNC_WAIT(n) __builtin_amdgcn_s_wait_asynccnt(n)
#else
#define ASYNC_WAIT(n) asm volatile("s_wait_asynccnt %0" :: "i"(n) : "memory")
#endif

// async global->LDS copy of 16 bytes/lane (GLOBAL_LOAD_ASYNC_TO_LDS_B128).
// ldsaddr: LDS byte offset (low 32 bits of a generic LDS pointer), per lane.
static __device__ __forceinline__ void async_copy_b128(unsigned ldsaddr,
                                                       const void* gaddr) {
  asm volatile("global_load_async_to_lds_b128 %0, %1, off"
               :: "v"(ldsaddr), "v"(gaddr) : "memory");
}

// ---------------------------------------------------------------------------
// f32 -> bf16 weight cast
// ---------------------------------------------------------------------------
__global__ void __launch_bounds__(256)
cast_f32_to_bf16(const float* __restrict__ src, __bf16* __restrict__ dst, int n) {
  int i = (blockIdx.x * 256 + threadIdx.x) * 4;
  if (i + 3 < n) {
    v4f x = *(const v4f*)(src + i);
    v4bf y = { (__bf16)x[0], (__bf16)x[1], (__bf16)x[2], (__bf16)x[3] };
    *(v4bf*)(dst + i) = y;
  }
}

// ---------------------------------------------------------------------------
// Embedding gather + cast: enc_bf[row, :] = bf16(embedding[inputs[row], :])
// ---------------------------------------------------------------------------
__global__ void __launch_bounds__(256)
gather_embed_bf16(const int* __restrict__ idx, const float* __restrict__ emb,
                  __bf16* __restrict__ out) {
  const int row = blockIdx.x;
  const int v   = idx[row];
  const float* src  = emb + (size_t)v * H_DIM;
  __bf16*      drow = out + (size_t)row * H_DIM;
  const int j = threadIdx.x * 4;
  v4f x = *(const v4f*)(src + j);
  v4bf y = { (__bf16)x[0], (__bf16)x[1], (__bf16)x[2], (__bf16)x[3] };
  *(v4bf*)(drow + j) = y;
}

// ---------------------------------------------------------------------------
// h0(bf16) = 0, step-barrier counter = 0
// ---------------------------------------------------------------------------
__global__ void __launch_bounds__(256)
init_state(__bf16* __restrict__ hb, int* __restrict__ bar) {
  const int i = blockIdx.x * 256 + threadIdx.x;
  hb[i] = (__bf16)0.0f;
  if (i == 0) *bar = 0;
}

// ---------------------------------------------------------------------------
// Big input projection: C[M, N_DIM] (f32) = A[M,K] (bf16) * W[N,K]^T (bf16)
// 256 threads = 8 waves; 128x128 tile.
// B tile staged with ASYNC global->LDS copies, triple-buffered with a
// write-ahead distance of 2 chunks so the DMA of chunk c+2 overlaps the
// WMMAs of chunk c+1; one workgroup barrier per chunk.
// ---------------------------------------------------------------------------
__global__ void __launch_bounds__(256)
gemm_bf16_wmma(const __bf16* __restrict__ A, const __bf16* __restrict__ W,
               float* __restrict__ C, int M) {
  __shared__ __attribute__((aligned(128))) __bf16 ldsB[3][128 * 32];  // 24 KB

  const int tid  = threadIdx.x;
  const int wave = tid >> 5;
  const int lane = tid & 31;
  const int l16  = lane & 15;
  const int lh   = lane >> 4;
  const int n0   = blockIdx.x * 128;
  const int m0   = blockIdx.y * 128 + wave * 16;

  int arow = m0 + l16;
  if (arow >= M) arow = M - 1;
  const __bf16* aptr = A + (size_t)arow * K_DIM + lh * 8;

  // staging coords: each thread stages 32 B of the B tile (2 async B128 ops)
  const int br = tid >> 1;                 // 0..127: row within the N tile
  const int bc = (tid & 1) * 16;           // 0 or 16: col within the K chunk
  const __bf16* wptr = W + (size_t)(n0 + br) * K_DIM + bc;
  const unsigned lbase = (unsigned)(uintptr_t)&ldsB[0][br * 32 + bc];

  const int NCHUNK = K_DIM / 32;           // 32 chunks of K
#define STAGE(c)                                                         \
  do {                                                                   \
    const __bf16* g = wptr + (c) * 32;                                   \
    unsigned l = lbase + ((c) % 3) * (128 * 32 * 2);                     \
    async_copy_b128(l,      (const void*)g);                             \
    async_copy_b128(l + 16, (const void*)(g + 8));                       \
  } while (0)

  STAGE(0);
  STAGE(1);

  v8f acc[8] = {};

  for (int c = 0; c < NCHUNK; ++c) {
    // async loads complete in order: allowing the 2 newest (chunk c+1) to be
    // outstanding guarantees chunk c has landed in LDS.
    if (c + 1 < NCHUNK) ASYNC_WAIT(2); else ASYNC_WAIT(0);
    __syncthreads();                       // chunk c visible to all waves

    const int k0 = c * 32;
    v16bf afrag = cat8(*(const v8bf*)(aptr + k0),
                       *(const v8bf*)(aptr + k0 + 16));
    const __bf16* lb = &ldsB[c % 3][0];
#pragma unroll
    for (int t = 0; t < 8; ++t) {
      const __bf16* bsrc = lb + (t * 16 + l16) * 32 + lh * 8;
      v16bf bfrag = cat8(*(const v8bf*)bsrc, *(const v8bf*)(bsrc + 16));
      acc[t] = __builtin_amdgcn_wmma_f32_16x16x32_bf16(
          false, afrag, false, bfrag, (short)0, acc[t], false, false);
    }

    if (c + 2 < NCHUNK) STAGE(c + 2);      // overlaps with next chunk's WMMAs
  }
#undef STAGE

#pragma unroll
  for (int t = 0; t < 8; ++t) {
    const int n = n0 + t * 16 + l16;
#pragma unroll
    for (int v = 0; v < 8; ++v) {
      const int m = m0 + lh * 8 + v;
      if (m < M) C[(size_t)m * N_DIM + n] = acc[t][v];
    }
  }
}

// ---------------------------------------------------------------------------
// Persistent fused GRU recurrence. One launch does all 512 steps.
//   grid = NWG_STEP WGs x 256 threads (8 waves).
//   WG owns h columns [col0, col0+64). Per step it computes the three 64x64
//   gate tiles hp[:, g*H + cols] with WMMA (K=1024 over bf16 h ping-pong
//   buffers), does the gate math in registers (the 3 gate accumulators for a
//   given (b,j) live in the same thread slot), keeps h(f32) resident in LDS
//   across all steps, writes ann + h(bf16), then crosses a device-scope
//   atomic barrier so every WG sees the full new h.
//   wave -> (mt = wave&3: rows 16*mt..; ch = wave>>2: cols 32*ch..)
// ---------------------------------------------------------------------------
__global__ void __launch_bounds__(256)
gru_recurrent_persistent(const __bf16* __restrict__ whh,    // [3H, K] bf16
                         const float*  __restrict__ xproj,  // [B, S, 3H]
                         const float*  __restrict__ b_ih,
                         const float*  __restrict__ b_hh,
                         __bf16* __restrict__ hb0,          // ping-pong h bf16
                         __bf16* __restrict__ hb1,
                         float*  __restrict__ ann,          // [B, S, H]
                         float*  __restrict__ hidden,       // [B, H]
                         int*    __restrict__ bar) {
  __shared__ float hlds[B_DIM * COLS_WG];   // 16 KB, persists across steps

  const int tid  = threadIdx.x;
  const int wave = tid >> 5;
  const int lane = tid & 31;
  const int l16  = lane & 15;
  const int lh   = lane >> 4;
  const int mt   = wave & 3;                // rows 16*mt .. 16*mt+15
  const int ch   = wave >> 2;               // cols 32*ch .. 32*ch+31 (local)
  const int col0 = blockIdx.x * COLS_WG;

  // h = 0
  for (int i = tid; i < B_DIM * COLS_WG; i += 256) hlds[i] = 0.0f;
  __syncthreads();

  // B row bases for the 6 (gate, subtile) output tiles of this wave
  const __bf16* bbase[6];
#pragma unroll
  for (int g = 0; g < 3; ++g)
#pragma unroll
    for (int t = 0; t < 2; ++t)
      bbase[g * 2 + t] =
          whh + (size_t)(g * H_DIM + col0 + ch * 32 + t * 16 + l16) * K_DIM + lh * 8;

  for (int s = 0; s < S_DIM; ++s) {
    const __bf16* hin  = (s & 1) ? hb1 : hb0;
    __bf16*       hout = (s & 1) ? hb0 : hb1;

    // ---- hp tiles: D = h(64xK) x whh_cols(Kx16) ----
    v8f acc[6] = {};
    const __bf16* aptr = hin + (size_t)(16 * mt + l16) * K_DIM + lh * 8;
    for (int k0 = 0; k0 < K_DIM; k0 += 32) {
      v16bf afrag = cat8(*(const v8bf*)(aptr + k0),
                         *(const v8bf*)(aptr + k0 + 16));
#pragma unroll
      for (int u = 0; u < 6; ++u) {
        const __bf16* bp = bbase[u] + k0;
        v16bf bfrag = cat8(*(const v8bf*)bp, *(const v8bf*)(bp + 16));
        acc[u] = __builtin_amdgcn_wmma_f32_16x16x32_bf16(
            false, afrag, false, bfrag, (short)0, acc[u], false, false);
      }
    }

    // ---- gate math, all register-local per (b, j) ----
#pragma unroll
    for (int t = 0; t < 2; ++t) {
      const int jl = ch * 32 + t * 16 + l16;        // local column
      const int j  = col0 + jl;                     // global column
      const float bihr = b_ih[j],             bhhr = b_hh[j];
      const float bihz = b_ih[H_DIM + j],     bhhz = b_hh[H_DIM + j];
      const float bihn = b_ih[2 * H_DIM + j], bhhn = b_hh[2 * H_DIM + j];
#pragma unroll
      for (int v = 0; v < 8; ++v) {
        const int b = 16 * mt + lh * 8 + v;
        const float* xrow = xproj + ((size_t)b * S_DIM + s) * N_DIM;
        float hr = acc[0 * 2 + t][v] + bhhr;
        float hz = acc[1 * 2 + t][v] + bhhz;
        float hn = acc[2 * 2 + t][v] + bhhn;
        float xr = xrow[j]             + bihr;
        float xz = xrow[H_DIM + j]     + bihz;
        float xn = xrow[2 * H_DIM + j] + bihn;
        float r = 1.0f / (1.0f + __expf(-(xr + hr)));
        float z = 1.0f / (1.0f + __expf(-(xz + hz)));
        float n = tanhf(xn + r * hn);
        float hprev = hlds[b * COLS_WG + jl];
        float h = (1.0f - z) * n + z * hprev;
        hlds[b * COLS_WG + jl] = h;                          // exclusive owner
        ann[((size_t)b * S_DIM + s) * H_DIM + j] = h;
        hout[(size_t)b * H_DIM + j] = (__bf16)h;
        if (s == S_DIM - 1) hidden[(size_t)b * H_DIM + j] = h;
      }
    }

    // ---- device-scope step barrier (monotonic counter) ----
    __threadfence();                 // make this thread's h stores visible
    __syncthreads();                 // whole WG done with step s
    if (tid == 0) {
      __hip_atomic_fetch_add(bar, 1, __ATOMIC_RELEASE, __HIP_MEMORY_SCOPE_AGENT);
      const int target = (s + 1) * NWG_STEP;
      while (__hip_atomic_load(bar, __ATOMIC_ACQUIRE, __HIP_MEMORY_SCOPE_AGENT) < target)
        __builtin_amdgcn_s_sleep(1);
    }
    __syncthreads();
    __threadfence();                 // acquire side: see all WGs' h stores
  }
}

// ---------------------------------------------------------------------------
extern "C" void kernel_launch(void* const* d_in, const int* in_sizes, int n_in,
                              void* d_out, int out_size, void* d_ws, size_t ws_size,
                              hipStream_t stream) {
  const int*   inputs = (const int*)  d_in[0];
  const float* emb    = (const float*)d_in[1];
  const float* w_ih   = (const float*)d_in[2];
  const float* w_hh   = (const float*)d_in[3];
  const float* b_ih   = (const float*)d_in[4];
  const float* b_hh   = (const float*)d_in[5];

  float* ann    = (float*)d_out;                        // [B, S, H]
  float* hidden = ann + (size_t)B_DIM * S_DIM * H_DIM;  // [B, H]

  // workspace carve-up
  char* p = (char*)d_ws;
  __bf16* enc_bf = (__bf16*)p; p += (size_t)B_DIM * S_DIM * H_DIM * 2;  // 67 MB
  __bf16* wih_bf = (__bf16*)p; p += (size_t)N_DIM * K_DIM * 2;         // 6.3 MB
  __bf16* whh_bf = (__bf16*)p; p += (size_t)N_DIM * K_DIM * 2;         // 6.3 MB
  float*  xproj  = (float*)p;  p += (size_t)B_DIM * S_DIM * N_DIM * 4; // 403 MB
  __bf16* hb0    = (__bf16*)p; p += (size_t)B_DIM * H_DIM * 2;
  __bf16* hb1    = (__bf16*)p; p += (size_t)B_DIM * H_DIM * 2;
  int*    bar    = (int*)p;    p += 256;

  const int wElems = N_DIM * K_DIM;
  cast_f32_to_bf16<<<wElems / (256 * 4), 256, 0, stream>>>(w_ih, wih_bf, wElems);
  cast_f32_to_bf16<<<wElems / (256 * 4), 256, 0, stream>>>(w_hh, whh_bf, wElems);
  gather_embed_bf16<<<B_DIM * S_DIM, 256, 0, stream>>>(inputs, emb, enc_bf);
  init_state<<<(B_DIM * H_DIM) / 256, 256, 0, stream>>>(hb0, bar);

  // Big input-projection GEMM: [32768,1024] x [1024,3072]
  dim3 gBig(N_DIM / 128, (B_DIM * S_DIM) / 128);
  gemm_bf16_wmma<<<gBig, 256, 0, stream>>>(enc_bf, wih_bf, xproj, B_DIM * S_DIM);

  // Whole recurrence in a single persistent launch
  gru_recurrent_persistent<<<NWG_STEP, 256, 0, stream>>>(
      whh_bf, xproj, b_ih, b_hh, hb0, hb1, ann, hidden, bar);
}
